// ParallelMamba_11046655885396
// MI455X (gfx1250) — matrix operand order repaired and checked by reading, not compiled, not run on hardware
//
#include <hip/hip_runtime.h>

// ---------------- problem constants ----------------
#define D_MODEL 1024
#define D_INNER 2048
#define N_STATE 16
#define DT_RANK 64
#define B_SZ 2
#define SEQ 1024
#define NROWS (B_SZ * SEQ) // 2048 token rows

// ---------------- WMMA types (CDNA5 gfx1250, wave32) ----------------
typedef __attribute__((ext_vector_type(16))) __bf16 v16bf;
typedef __attribute__((ext_vector_type(8)))  float  v8f;

union FragAB { v16bf v; unsigned int u[8]; };

enum { ACT_NONE = 0, ACT_SOFTPLUS = 1 };

__device__ __forceinline__ unsigned short f32_to_bf16(float f) {
  union { float f; unsigned int u; } x; x.f = f;
  unsigned int u = x.u;
  unsigned int r = u + 0x7FFFu + ((u >> 16) & 1u); // round-to-nearest-even
  return (unsigned short)(r >> 16);
}

// 16-byte-chunk XOR swizzle: rows are 64B (4 chunks); decorrelate both row%4
// and row/4 so b128 LDS reads across 16 rows land in distinct bank groups.
__device__ __forceinline__ int swz(int row) { return (row & 3) ^ ((row >> 2) & 3); }

// ---- CDNA5 async copy global->LDS (ASYNCcnt-tracked) ----
__device__ __forceinline__ void async_copy16(unsigned lds_off, const void* g) {
  unsigned long long ga = (unsigned long long)(uintptr_t)g;
  asm volatile("global_load_async_to_lds_b128 %0, %1, off"
               :: "v"(lds_off), "v"(ga) : "memory");
}
__device__ __forceinline__ void wait_async0() { asm volatile("s_wait_asynccnt 0x0" ::: "memory"); }
__device__ __forceinline__ void wait_async4() { asm volatile("s_wait_asynccnt 0x4" ::: "memory"); }

// ---------------------------------------------------------------------------
// GEMM: C[M,N] = act(A * B + bias).  A pre-packed bf16 [M][K] row-major,
// B pre-packed bf16 TRANSPOSED [Npad][K] (Npad = ceil(N/128)*128, zero-filled).
// Block 256 thr = 8 waves, tile 128M x 128N; each wave: 2 M-tiles x 4 N-tiles
// = 8 WMMA per K-step (12 ds_load_b128 per 8 WMMAs).  Double-buffered LDS
// staging via global_load_async_to_lds_b128 (4 b128 copies/thread/stage).
// Requires M%128==0, K%32==0, K>=64 (T>=2).
// ---------------------------------------------------------------------------
#define STAGE_BYTES 16384   // A: 128 rows * 64B = 8KB, B: 128 rows * 64B = 8KB

__global__ __launch_bounds__(256) void gemm_bf16_async(
    const unsigned short* __restrict__ Ab,   // [M][K] bf16
    const unsigned short* __restrict__ Bt,   // [Npad][K] bf16 (transposed)
    const float* __restrict__ bias,
    float* __restrict__ C, int ldc,
    int M, int N, int Kdim, int act)
{
  __shared__ __align__(16) unsigned short lds[2 * STAGE_BYTES / 2]; // 32KB

  const int tid  = threadIdx.x;
  const int lane = tid & 31;
  const int wave = tid >> 5;
  const int half = lane >> 4;
  const int l15  = lane & 15;
  const int mg   = wave & 3;   // m-pair of this wave: m-tiles mg*2, mg*2+1
  const int ng   = wave >> 2;  // n-group (0..1): n-tiles ng*4 .. ng*4+3
  const int m0 = blockIdx.y * 128;
  const int n0 = blockIdx.x * 128;

  const unsigned lds_base = (unsigned)(uintptr_t)lds; // LDS byte offset

  // per-thread copy assignment: A and B each 128 rows * 4 chunks = 512 chunks;
  // each thread copies chunks {tid, tid+256} of A and of B (16B per chunk).
  const int r0 = tid >> 2,           c0 = tid & 3;
  const int r1 = (tid + 256) >> 2,   c1 = tid & 3;

  const size_t a_rb0 = (size_t)(m0 + r0) * Kdim;
  const size_t a_rb1 = (size_t)(m0 + r1) * Kdim;
  const size_t b_rb0 = (size_t)(n0 + r0) * Kdim;
  const size_t b_rb1 = (size_t)(n0 + r1) * Kdim;
  const unsigned a_d0 = r0 * 64 + ((c0 ^ swz(r0)) << 4);
  const unsigned a_d1 = r1 * 64 + ((c1 ^ swz(r1)) << 4);
  const unsigned b_d0 = 8192 + a_d0;   // same row/chunk pattern, B half of stage
  const unsigned b_d1 = 8192 + a_d1;

  v8f acc[2][4] = {};

  // prologue: stage 0
  async_copy16(lds_base + a_d0, (const char*)(Ab + a_rb0) + c0 * 16);
  async_copy16(lds_base + a_d1, (const char*)(Ab + a_rb1) + c1 * 16);
  async_copy16(lds_base + b_d0, (const char*)(Bt + b_rb0) + c0 * 16);
  async_copy16(lds_base + b_d1, (const char*)(Bt + b_rb1) + c1 * 16);

  const int T = Kdim >> 5;
  const int sw = swz(l15);

  for (int t = 0; t < T; ++t) {
    const int stage = t & 1;
    if (t + 1 < T) {
      // issue next stage into the other buffer (its last readers were fenced
      // by the trailing barrier of iteration t-1)
      const unsigned sb = (stage ^ 1) * STAGE_BYTES;
      const int k0 = (t + 1) << 5;
      async_copy16(lds_base + sb + a_d0, (const char*)(Ab + a_rb0 + k0) + c0 * 16);
      async_copy16(lds_base + sb + a_d1, (const char*)(Ab + a_rb1 + k0) + c1 * 16);
      async_copy16(lds_base + sb + b_d0, (const char*)(Bt + b_rb0 + k0) + c0 * 16);
      async_copy16(lds_base + sb + b_d1, (const char*)(Bt + b_rb1 + k0) + c1 * 16);
      wait_async4();   // retire stage t's 4 copies; stage t+1 stays in flight
    } else {
      wait_async0();
    }
    __syncthreads();

    const unsigned short* sp = lds + stage * (STAGE_BYTES / 2);
    const unsigned int* a32 = (const unsigned int*)sp;          // 128 rows * 16 dw
    const unsigned int* b32 = (const unsigned int*)(sp + 4096); // 128 rows * 16 dw

    // A fragments (2 m-tiles): lane holds row m; e -> k=(e<8?0:16)+half*8+(e&7)
    FragAB fa[2];
#pragma unroll
    for (int i = 0; i < 2; ++i) {
      const int rowoff = ((mg * 2 + i) * 16 + l15) * 16;
      const int ca = (half ^ sw) << 2, cb = ((2 + half) ^ sw) << 2;
#pragma unroll
      for (int j = 0; j < 4; ++j) {
        fa[i].u[j]     = a32[rowoff + ca + j];
        fa[i].u[4 + j] = a32[rowoff + cb + j];
      }
    }
    // 4 B fragments x 2 A fragments = 8 WMMAs; B: lane holds col n; e -> k=(e<8?0:16)+half*8+(e&7)
#pragma unroll
    for (int j = 0; j < 4; ++j) {
      const int rowoff = ((ng * 4 + j) * 16 + l15) * 16;
      FragAB fb;
#pragma unroll
      for (int e = 0; e < 8; ++e) {
        const int c = (half + 2 * (e >> 2)) ^ sw;
        fb.u[e] = b32[rowoff + (c << 2) + (e & 3)];
      }
      acc[0][j] = __builtin_amdgcn_wmma_f32_16x16x32_bf16(
          false, fa[0].v, false, fb.v, (short)0, acc[0][j], false, false);
      acc[1][j] = __builtin_amdgcn_wmma_f32_16x16x32_bf16(
          false, fa[1].v, false, fb.v, (short)0, acc[1][j], false, false);
    }
    __syncthreads();
  }

  // epilogue: D layout per 16x16 tile: VGPR r -> M = half*8 + r, N = l15
#pragma unroll
  for (int i = 0; i < 2; ++i) {
#pragma unroll
    for (int j = 0; j < 4; ++j) {
      const int n = n0 + (ng * 4 + j) * 16 + l15;
      if (n < N) {
        const float bv = bias ? bias[n] : 0.f;
        const int mrow = m0 + (mg * 2 + i) * 16 + half * 8;
#pragma unroll
        for (int r = 0; r < 8; ++r) {
          float v = acc[i][j][r] + bv;
          if (act == ACT_SOFTPLUS) v = (v > 20.f) ? v : log1pf(__expf(v));
          C[(size_t)(mrow + r) * ldc + n] = v;
        }
      }
    }
  }
}

// ---------------------------------------------------------------------------
// Pack helpers: f32 -> bf16 (A row-major, possibly strided; B transposed+padded)
// ---------------------------------------------------------------------------
__global__ __launch_bounds__(256) void pack_a_bf16(
    const float* __restrict__ src, int ld, int cols,
    unsigned short* __restrict__ dst, int total)
{
  int i = blockIdx.x * 256 + threadIdx.x;
  if (i < total) {
    int r = i / cols, c = i - r * cols;
    dst[i] = f32_to_bf16(src[(size_t)r * ld + c]);
  }
}

__global__ __launch_bounds__(256) void pack_bt_bf16(
    const float* __restrict__ src, int K, int N, int Npad,
    unsigned short* __restrict__ dst)
{
  int i = blockIdx.x * 256 + threadIdx.x; // over Npad*K
  if (i < Npad * K) {
    int n = i / K, k = i - n * K;
    dst[i] = (n < N) ? f32_to_bf16(src[(size_t)k * N + n]) : (unsigned short)0;
  }
}

// ---------------------------------------------------------------------------
// Causal depthwise conv1d (K=3) + bias + SiLU; emits f32 u and bf16 ub
// ---------------------------------------------------------------------------
__global__ __launch_bounds__(256) void conv_silu_kernel(
    const float* __restrict__ xr, const float* __restrict__ conv_w,
    const float* __restrict__ conv_b, float* __restrict__ u,
    unsigned short* __restrict__ ub)
{
  int idx = blockIdx.x * 256 + threadIdx.x; // over NROWS * D_INNER
  int row = idx >> 11;
  int d   = idx & 2047;
  int l   = row & (SEQ - 1);
  float w0 = conv_w[d * 3 + 0], w1 = conv_w[d * 3 + 1], w2 = conv_w[d * 3 + 2];
  float a = conv_b[d] + w2 * xr[(size_t)row * 4096 + d];
  if (l >= 1) a += w1 * xr[(size_t)(row - 1) * 4096 + d];
  if (l >= 2) a += w0 * xr[(size_t)(row - 2) * 4096 + d];
  float s = a / (1.f + __expf(-a)); // silu
  u[(size_t)row * 2048 + d] = s;
  ub[(size_t)row * 2048 + d] = f32_to_bf16(s);
}

// ---------------------------------------------------------------------------
// Selective scan: one thread per (b,d) channel; h[16] in VGPRs, v_exp_f32.
// xp rows hold [dt(64) | B(16) | C(16)].
// ---------------------------------------------------------------------------
__global__ __launch_bounds__(256) void scan_kernel(
    const float* __restrict__ u, const float* __restrict__ delta,
    const float* __restrict__ xp, const float* __restrict__ A_log,
    float* __restrict__ y)
{
  int ch = blockIdx.x * 256 + threadIdx.x;  // 0..4095
  int b = ch >> 11;
  int d = ch & 2047;
  float An[16], h[16];
#pragma unroll
  for (int n = 0; n < 16; ++n) { An[n] = -__expf(A_log[d * 16 + n]); h[n] = 0.f; }
  for (int l = 0; l < SEQ; ++l) {
    int row = b * SEQ + l;
    float dl = delta[(size_t)row * 2048 + d];
    float ul = u[(size_t)row * 2048 + d];
    float du = dl * ul;
    const float* xrow = xp + (size_t)row * 96;
    float yv = 0.f;
#pragma unroll
    for (int n = 0; n < 16; ++n) {
      float dA = __expf(dl * An[n]);             // exp(delta * A)
      h[n] = fmaf(dA, h[n], du * xrow[64 + n]);  // h = dA*h + delta*B*u
      yv   = fmaf(h[n], xrow[80 + n], yv);       // y += h * C
    }
    y[(size_t)row * 2048 + d] = yv;
  }
}

// ---------------------------------------------------------------------------
// Gate: ymb = bf16( y * silu(res) ), res = xr[:, 2048:4096]
// ---------------------------------------------------------------------------
__global__ __launch_bounds__(256) void gate_kernel(
    const float* __restrict__ y, const float* __restrict__ xr,
    unsigned short* __restrict__ ymb)
{
  int idx = blockIdx.x * 256 + threadIdx.x;
  int row = idx >> 11;
  int d   = idx & 2047;
  float r = xr[(size_t)row * 4096 + 2048 + d];
  float s = r / (1.f + __expf(-r));
  ymb[idx] = f32_to_bf16(y[idx] * s);
}

// ---------------------------------------------------------------------------
extern "C" void kernel_launch(void* const* d_in, const int* in_sizes, int n_in,
                              void* d_out, int out_size, void* d_ws, size_t ws_size,
                              hipStream_t stream) {
  const float* x      = (const float*)d_in[0];   // [2,1024,1024]
  const float* A_log  = (const float*)d_in[1];   // [2048,16]
  const float* W_in   = (const float*)d_in[2];   // [1024,4096]
  const float* b_in   = (const float*)d_in[3];   // [4096]
  const float* conv_w = (const float*)d_in[4];   // [2048,1,3]
  const float* conv_b = (const float*)d_in[5];   // [2048]
  const float* W_x    = (const float*)d_in[6];   // [2048,96]
  const float* W_dt   = (const float*)d_in[7];   // [64,2048]
  const float* b_dt   = (const float*)d_in[8];   // [2048]
  const float* W_out  = (const float*)d_in[9];   // [2048,1024]
  const float* b_out  = (const float*)d_in[10];  // [1024]
  float* out = (float*)d_out;                    // [2,1024,1024] f32

  char* ws = (char*)d_ws;
  float* xr    = (float*)(ws + 0);            // [2048,4096] f32  32 MB
  float* u     = (float*)(ws + 33554432);     // [2048,2048] f32  16 MB
  float* xp    = (float*)(ws + 50331648);     // [2048,96]   f32
  float* delta = (float*)(ws + 51118080);     // [2048,2048] f32  16 MB
  float* y     = (float*)(ws + 67895296);     // [2048,2048] f32  16 MB
  unsigned short* xb    = (unsigned short*)(ws + 84672512);  // [2048,1024]  4 MB
  unsigned short* ub    = (unsigned short*)(ws + 88866816);  // [2048,2048]  8 MB
  unsigned short* ymb   = (unsigned short*)(ws + 97255424);  // [2048,2048]  8 MB
  unsigned short* dtb   = (unsigned short*)(ws + 105644032); // [2048,64]
  unsigned short* WinT  = (unsigned short*)(ws + 105906176); // [4096,1024]  8 MB
  unsigned short* WxT   = (unsigned short*)(ws + 114294784); // [128,2048]  (Npad=128)
  unsigned short* WdtT  = (unsigned short*)(ws + 114819072); // [2048,64]
  unsigned short* WoutT = (unsigned short*)(ws + 115081216); // [1024,2048]  4 MB

  // ---- pack operands to bf16 (A row-major; B transposed [Npad][K]) ----
  pack_a_bf16<<<(2048 * 1024) / 256, 256, 0, stream>>>(x, 1024, 1024, xb, 2048 * 1024);
  pack_bt_bf16<<<(4096 * 1024) / 256, 256, 0, stream>>>(W_in, 1024, 4096, 4096, WinT);
  pack_bt_bf16<<<(128 * 2048) / 256, 256, 0, stream>>>(W_x, 2048, 96, 128, WxT);
  pack_bt_bf16<<<(2048 * 64) / 256, 256, 0, stream>>>(W_dt, 64, 2048, 2048, WdtT);
  pack_bt_bf16<<<(1024 * 2048) / 256, 256, 0, stream>>>(W_out, 2048, 1024, 1024, WoutT);

  // 1) xr = x @ W_in + b_in        (M=2048, K=1024, N=4096)
  gemm_bf16_async<<<dim3(32, 16), 256, 0, stream>>>(
      xb, WinT, b_in, xr, 4096, 2048, 4096, 1024, ACT_NONE);

  // 2) u = silu(causal_conv(u_raw) + conv_b)   (also bf16 copy)
  conv_silu_kernel<<<(NROWS * D_INNER) / 256, 256, 0, stream>>>(xr, conv_w, conv_b, u, ub);

  // 3) xp = u @ W_x                (M=2048, K=2048, N=96)
  gemm_bf16_async<<<dim3(1, 16), 256, 0, stream>>>(
      ub, WxT, nullptr, xp, 96, 2048, 96, 2048, ACT_NONE);

  // 3b) dt view -> packed bf16
  pack_a_bf16<<<(2048 * 64) / 256, 256, 0, stream>>>(xp, 96, 64, dtb, 2048 * 64);

  // 4) delta = softplus(dt @ W_dt + b_dt)      (M=2048, K=64, N=2048)
  gemm_bf16_async<<<dim3(16, 16), 256, 0, stream>>>(
      dtb, WdtT, b_dt, delta, 2048, 2048, 2048, 64, ACT_SOFTPLUS);

  // 5) selective scan -> y
  scan_kernel<<<(B_SZ * D_INNER) / 256, 256, 0, stream>>>(u, delta, xp, A_log, y);

  // 6) ymb = bf16(y * silu(res))
  gate_kernel<<<(NROWS * D_INNER) / 256, 256, 0, stream>>>(y, xr, ymb);

  // 7) out = ym @ W_out + b_out    (M=2048, K=2048, N=1024)
  gemm_bf16_async<<<dim3(8, 16), 256, 0, stream>>>(
      ymb, WoutT, b_out, out, 1024, 2048, 1024, 2048, ACT_NONE);
}
